// VanillaAttention_23175643529613
// MI455X (gfx1250) — matrix operand order repaired and checked by reading, not compiled, not run on hardware
//
#include <hip/hip_runtime.h>
#include <hip/hip_bf16.h>

// ---------------------------------------------------------------------------
// Causal self-attention block for MI455X (gfx1250, wave32, WMMA 16x16x32 bf16)
//   qkv = x @ W_attn + b_attn ; flash-attention per head ; out = y @ W_proj + b
// Matmuls on v_wmma_f32_16x16x32_bf16. GEMM tiles are staged by the Tensor
// Data Mover with a double-buffered software pipeline (tensor_load_to_lds for
// tile k+1 overlaps the WMMAs on tile k; s_wait_tensorcnt + one barrier per
// k-step). fp32 softmax.
// ---------------------------------------------------------------------------

typedef __bf16 bf16;
typedef __attribute__((ext_vector_type(8)))  bf16  bf16x8;
typedef __attribute__((ext_vector_type(16))) bf16  bf16x16;
typedef __attribute__((ext_vector_type(8)))  float f32x8;
typedef __attribute__((ext_vector_type(4)))  unsigned u32x4;
typedef __attribute__((ext_vector_type(8)))  int   i32x8;
typedef __attribute__((ext_vector_type(4)))  int   i32x4;

#define B_  4
#define T_  2048
#define C_  1024
#define NH_ 16
#define HD_ 64
#define M_  (B_ * T_)      // 8192 rows
#define SR_ (3 * C_)       // qkv row stride: 3072

static __device__ __forceinline__ bf16x16 cat8(bf16x8 lo, bf16x8 hi) {
  return __builtin_shufflevector(lo, hi, 0,1,2,3,4,5,6,7,8,9,10,11,12,13,14,15);
}

static __device__ __forceinline__ f32x8 wmma_bf16(bf16x16 a, bf16x16 b, f32x8 c) {
  // (neg_a, A, neg_b, B, c_mod, C, reuse_a, reuse_b)
  return __builtin_amdgcn_wmma_f32_16x16x32_bf16(false, a, false, b, (short)0, c,
                                                 false, false);
}

// ---------------------------------------------------------------------------
// TDM: 2D tile load Global -> LDS via tensor descriptor (ISA 08_async_tensor).
// D# group0: count=1 | lds_addr | global_addr(57b) | type=2
// D# group1: data_size=2B, tensor_dim0/1, tile_dim0/1, tensor_dim0_stride
// Issued once per (designated) wave; EXEC is ignored by TDM; completion is
// tracked with TENSORcnt.
// ---------------------------------------------------------------------------
static __device__ __forceinline__ void tdm_load_2d_bf16(
    const bf16* gsrc,          // global address of tile start
    unsigned lds_byte_addr,    // LDS byte offset of destination
    int tile_d0,               // tile width  (elements, contiguous dim)
    int tile_d1,               // tile height (rows)
    int stride_elems,          // tensor dim0 stride (elements)
    int tensor_d0, int tensor_d1) {
  unsigned long long ga = (unsigned long long)(size_t)gsrc;
  u32x4 g0;
  g0[0] = 1u;                                            // count=1 (valid), user
  g0[1] = lds_byte_addr;                                 // lds_addr
  g0[2] = (unsigned)(ga & 0xFFFFFFFFu);                  // global_addr[31:0]
  g0[3] = (unsigned)((ga >> 32) & 0x1FFFFFFu) | (2u << 30);  // [56:32] | type=2
  i32x8 g1;
  g1[0] = (int)(1u << 16);                               // data_size = 2 bytes
  g1[1] = (int)(((unsigned)tensor_d0 & 0xFFFFu) << 16);  // tensor_dim0[15:0]
  g1[2] = (int)(((unsigned)tensor_d0 >> 16) |
                (((unsigned)tensor_d1 & 0xFFFFu) << 16));// td0[31:16] | td1[15:0]
  g1[3] = (int)(((unsigned)tensor_d1 >> 16) |
                ((unsigned)tile_d0 << 16));              // td1[31:16] | tile_dim0
  g1[4] = (int)(unsigned)tile_d1;                        // tile_dim1 | tile_dim2=0
  g1[5] = (int)(unsigned)stride_elems;                   // tensor_dim0_stride lo
  g1[6] = 0;                                             // stride0 hi | stride1 lo
  g1[7] = 0;                                             // stride1 hi
  i32x4 z4; z4[0] = z4[1] = z4[2] = z4[3] = 0;
#if __clang_major__ >= 23
  i32x8 z8; for (int i = 0; i < 8; ++i) z8[i] = 0;
  __builtin_amdgcn_tensor_load_to_lds(g0, g1, z4, z4, z8, 0);
#else
  __builtin_amdgcn_tensor_load_to_lds(g0, g1, z4, z4, 0);
#endif
}

// ---------------------------------------------------------------------------
// fp32 -> bf16 cast (row-major copy)
// ---------------------------------------------------------------------------
__global__ void cast_bf16_kernel(const float* __restrict__ src,
                                 bf16* __restrict__ dst, int n) {
  for (int i = blockIdx.x * blockDim.x + threadIdx.x; i < n;
       i += gridDim.x * blockDim.x)
    dst[i] = (bf16)src[i];
}

// fp32 KxN -> bf16 NxK (transpose so GEMM B-operand has contiguous K)
__global__ void transpose_cast_bf16_kernel(const float* __restrict__ src,
                                           bf16* __restrict__ dst,
                                           int K, int N) {
  int n = K * N;
  for (int i = blockIdx.x * blockDim.x + threadIdx.x; i < n;
       i += gridDim.x * blockDim.x) {
    int k = i / N;
    int c = i - k * N;
    dst[(size_t)c * K + k] = (bf16)src[i];
  }
}

// ---------------------------------------------------------------------------
// NT GEMM: C[M,N] = A[M,K] * Bt[N,K]^T + bias ;  A,Bt bf16, accum fp32.
// Block tile 128x128, 8 waves (2x4), 64x32 per wave, BK=64.
// Double-buffered TDM pipeline: DMA of tile k+1 overlaps WMMAs on tile k.
// ---------------------------------------------------------------------------
#define BM 128
#define BN 128
#define BK 64

__global__ __launch_bounds__(256) void gemm_bf16_nt_kernel(
    const bf16* __restrict__ A,    // M x K row-major
    const bf16* __restrict__ Bt,   // N x K row-major (pre-transposed weights)
    const float* __restrict__ bias,// N (may be null)
    float* __restrict__ Cf,        // fp32 out (may be null)
    bf16* __restrict__ Cb,         // bf16 out (may be null)
    int M, int N, int K) {
  __shared__ bf16 As[2][BM][BK];   // 2 x 16 KB
  __shared__ bf16 Bs[2][BN][BK];   // 2 x 16 KB

  const int tid  = threadIdx.x;
  const int wave = tid >> 5;
  const int lane = tid & 31;
  const int l15  = lane & 15;
  const int lhi  = lane >> 4;
  const int mblk = blockIdx.y * BM;
  const int nblk = blockIdx.x * BN;
  const int wm   = (wave >> 2) * 64;   // wave row origin inside block tile
  const int wn   = (wave & 3) * 32;    // wave col origin inside block tile

  // LDS byte addresses (generic LDS pointer low 32 bits == LDS offset)
  const unsigned lds_as[2] = {(unsigned)(size_t)&As[0][0][0],
                              (unsigned)(size_t)&As[1][0][0]};
  const unsigned lds_bs[2] = {(unsigned)(size_t)&Bs[0][0][0],
                              (unsigned)(size_t)&Bs[1][0][0]};

  const bf16* arow = &A[(size_t)mblk * K];
  const bf16* brow = &Bt[(size_t)nblk * K];

  f32x8 acc[4][2];
  for (int i = 0; i < 4; ++i)
    for (int j = 0; j < 2; ++j)
      for (int r = 0; r < 8; ++r) acc[i][j][r] = 0.0f;

  // Prologue: kick off the DMA for k-tile 0 into buffer 0.
  if (wave == 0)      tdm_load_2d_bf16(arow, lds_as[0], BK, BM, K, K, M);
  else if (wave == 1) tdm_load_2d_bf16(brow, lds_bs[0], BK, BN, K, K, N);

  int buf = 0;
  for (int k0 = 0; k0 < K; k0 += BK, buf ^= 1) {
    if (wave < 2) __builtin_amdgcn_s_wait_tensorcnt(0);  // buffer `buf` landed
    __syncthreads();  // publish tile; also: everyone done reading buf^1

    // Issue the DMA for the next k-tile into the other buffer; it runs
    // concurrently with the WMMAs below.
    if (k0 + BK < K) {
      if (wave == 0)
        tdm_load_2d_bf16(arow + k0 + BK, lds_as[buf ^ 1], BK, BM, K, K, M);
      else if (wave == 1)
        tdm_load_2d_bf16(brow + k0 + BK, lds_bs[buf ^ 1], BK, BN, K, K, N);
    }

    for (int kk = 0; kk < BK; kk += 32) {
      bf16x16 af[4];
      bf16x16 bfm[2];
      // A fragment (16x32): lanes 0-15 row m, K 0-7 & 16-23; lanes 16-31 K 8-15 & 24-31
      for (int i = 0; i < 4; ++i) {
        const bf16* p = &As[buf][wm + i * 16 + l15][kk + lhi * 8];
        af[i] = cat8(*(const bf16x8*)p, *(const bf16x8*)(p + 16));
      }
      // B fragment (32x16): lanes 0-15 col n, K 0-7 & 16-23; lanes 16-31 K 8-15 & 24-31
      for (int j = 0; j < 2; ++j) {
        const bf16* p = &Bs[buf][wn + j * 16 + l15][kk + lhi * 8];
        bfm[j] = cat8(*(const bf16x8*)p, *(const bf16x8*)(p + 16));
      }
      for (int i = 0; i < 4; ++i)
        for (int j = 0; j < 2; ++j)
          acc[i][j] = wmma_bf16(af[i], bfm[j], acc[i][j]);
    }
  }

  // Epilogue: C/D layout — VGPR r holds row (r + 8*lane_hi), col = lane&15.
  for (int i = 0; i < 4; ++i) {
    for (int j = 0; j < 2; ++j) {
      int ncol = nblk + wn + j * 16 + l15;
      float bv = bias ? bias[ncol] : 0.0f;
      for (int r = 0; r < 8; ++r) {
        int mrow = mblk + wm + i * 16 + r + lhi * 8;
        float v  = acc[i][j][r] + bv;
        if (Cf) Cf[(size_t)mrow * N + ncol] = v;
        if (Cb) Cb[(size_t)mrow * N + ncol] = (bf16)v;
      }
    }
  }
}

// ---------------------------------------------------------------------------
// Flash attention (causal). One block per (b, h, 128 q-rows); each of the 8
// waves owns 16 q-rows -> loop bounds are wave-uniform (EXEC all-1s at WMMA).
// 32-key tiles: S = Q K^T via 4 WMMAs, fp32 online softmax, P·V via 4 WMMAs.
// P is moved C-layout -> A-layout and V is transposed through per-wave LDS
// (per-wave LDS ops are in-order, so no block barrier in the divergent loop).
// ---------------------------------------------------------------------------
__global__ __launch_bounds__(256) void flash_attn_kernel(
    const bf16* __restrict__ qkv,  // (B*T) x 3C bf16
    bf16* __restrict__ Y) {        // (B*T) x C  bf16
  __shared__ bf16 Pls[8][16][32];  // per-wave P scratch   (8 KB)
  __shared__ bf16 Vls[8][64][32];  // per-wave V^T scratch (32 KB): [d][kk]

  const int wave = threadIdx.x >> 5;
  const int lane = threadIdx.x & 31;
  const int l15  = lane & 15;
  const int lhi  = lane >> 4;

  const int qt = blockIdx.x;      // 0..15
  const int h  = blockIdx.y;      // 0..15
  const int b  = blockIdx.z;      // 0..3

  const int qbase = qt * 128 + wave * 16;
  const bf16* qp = qkv + (size_t)(b * T_) * SR_ + h * HD_;
  const bf16* kp = qp + C_;
  const bf16* vp = qp + 2 * C_;

  // Q fragments for this wave's 16 rows (head dim 64 -> two 16x32 A-frags).
  bf16x16 qf[2];
  {
    const bf16* base = qp + (size_t)(qbase + l15) * SR_;
    for (int c = 0; c < 2; ++c) {
      const bf16* p = base + c * 32 + lhi * 8;
      qf[c] = cat8(*(const bf16x8*)p, *(const bf16x8*)(p + 16));
    }
  }

  float rmax[8], rsum[8];
  f32x8 oacc[4];
  for (int r = 0; r < 8; ++r) { rmax[r] = -1e30f; rsum[r] = 0.0f; }
  for (int t = 0; t < 4; ++t)
    for (int r = 0; r < 8; ++r) oacc[t][r] = 0.0f;

  const float scale = 0.125f;  // 1/sqrt(64)
  const int qmax = qbase + 15;

  for (int j0 = 0; j0 <= qmax; j0 += 32) {
    // Prefetch next key tile rows while we compute on this one.
    if (j0 + 32 <= qmax) {
      __builtin_prefetch(kp + (size_t)(j0 + 32 + lane) * SR_, 0, 1);
      __builtin_prefetch(vp + (size_t)(j0 + 32 + lane) * SR_, 0, 1);
    }

    // ---- S = Q K^T for 32 keys (two 16-key column groups) ----
    f32x8 s[2];
    for (int nc = 0; nc < 2; ++nc) {
      const bf16* kb = kp + (size_t)(j0 + nc * 16 + l15) * SR_ + lhi * 8;
      bf16x16 kf0 = cat8(*(const bf16x8*)kb,        *(const bf16x8*)(kb + 16));
      bf16x16 kf1 = cat8(*(const bf16x8*)(kb + 32), *(const bf16x8*)(kb + 48));
      f32x8 c;
      for (int r = 0; r < 8; ++r) c[r] = 0.0f;
      c = wmma_bf16(qf[0], kf0, c);
      c = wmma_bf16(qf[1], kf1, c);
      s[nc] = c;
    }

    // ---- fp32 online softmax (rows live across a 16-lane half) ----
    for (int r = 0; r < 8; ++r) {
      int   qrow = qbase + r + lhi * 8;
      int   key0 = j0 + l15;
      int   key1 = j0 + 16 + l15;
      float s0 = (key0 <= qrow) ? s[0][r] * scale : -1e30f;
      float s1 = (key1 <= qrow) ? s[1][r] * scale : -1e30f;
      float mx = fmaxf(s0, s1);
      for (int off = 8; off; off >>= 1) mx = fmaxf(mx, __shfl_xor(mx, off, 32));
      float nm   = fmaxf(rmax[r], mx);
      float corr = __expf(rmax[r] - nm);
      float p0   = __expf(s0 - nm);
      float p1   = __expf(s1 - nm);
      float ps   = p0 + p1;
      for (int off = 8; off; off >>= 1) ps += __shfl_xor(ps, off, 32);
      rsum[r] = rsum[r] * corr + ps;
      rmax[r] = nm;
      for (int t = 0; t < 4; ++t) oacc[t][r] *= corr;
      // P: C-layout -> LDS (row m = r + 8*lane_hi, col = key-in-tile)
      int m = r + lhi * 8;
      Pls[wave][m][l15]      = (bf16)p0;
      Pls[wave][m][16 + l15] = (bf16)p1;
    }

    // ---- stage V tile transposed: lane handles key row j0+lane ----
    {
      const bf16* vb = vp + (size_t)(j0 + lane) * SR_;
      for (int c = 0; c < 8; ++c) {
        bf16x8 vv = *(const bf16x8*)(vb + c * 8);
        for (int e = 0; e < 8; ++e) Vls[wave][c * 8 + e][lane] = vv[e];
      }
    }
    asm volatile("" ::: "memory");  // compiler barrier; HW LDS is in-order/wave

    // ---- P as A-fragment ----
    bf16x16 pf;
    {
      const bf16* p = &Pls[wave][l15][lhi * 8];
      pf = cat8(*(const bf16x8*)p, *(const bf16x8*)(p + 16));
    }
    // ---- O += P·V : four 16-wide d chunks ----
    for (int t = 0; t < 4; ++t) {
      const bf16* vb2 = &Vls[wave][t * 16 + l15][lhi * 8];
      bf16x16 vf = cat8(*(const bf16x8*)vb2, *(const bf16x8*)(vb2 + 16));
      oacc[t] = wmma_bf16(pf, vf, oacc[t]);
    }
  }

  // ---- normalize and emit y (bf16, (B*T) x C) ----
  for (int r = 0; r < 8; ++r) {
    float inv = 1.0f / rsum[r];
    int   m   = qbase + r + lhi * 8;
    for (int t = 0; t < 4; ++t) {
      int col = h * HD_ + t * 16 + l15;
      Y[(size_t)(b * T_ + m) * C_ + col] = (bf16)(oacc[t][r] * inv);
    }
  }
}

// ---------------------------------------------------------------------------
// Host-side orchestration (graph-capture safe: kernel launches only)
// ---------------------------------------------------------------------------
extern "C" void kernel_launch(void* const* d_in, const int* in_sizes, int n_in,
                              void* d_out, int out_size, void* d_ws, size_t ws_size,
                              hipStream_t stream) {
  const float* x      = (const float*)d_in[0];  // (B,T,C)
  const float* W_attn = (const float*)d_in[1];  // (C, 3C)
  const float* b_attn = (const float*)d_in[2];  // (3C)
  const float* W_proj = (const float*)d_in[3];  // (C, C)
  const float* b_proj = (const float*)d_in[4];  // (C)
  float* out = (float*)d_out;                   // (B,T,C) fp32

  char* ws = (char*)d_ws;
  bf16* xb   = (bf16*)(ws);                         // 8192*1024   = 16 MB
  bf16* WaT  = (bf16*)(ws + 16777216);              // 3072*1024   =  6 MB
  bf16* WpT  = (bf16*)(ws + 23068672);              // 1024*1024   =  2 MB
  bf16* qkvb = (bf16*)(ws + 25165824);              // 8192*3072   = 48 MB
  bf16* yb   = (bf16*)(ws + 75497472);              // 8192*1024   = 16 MB
  (void)ws_size; (void)n_in; (void)in_sizes; (void)out_size;

  // 1) downcast activations / transpose+downcast weights for NT WMMA GEMMs
  cast_bf16_kernel<<<2048, 256, 0, stream>>>(x, xb, M_ * C_);
  transpose_cast_bf16_kernel<<<2048, 256, 0, stream>>>(W_attn, WaT, C_, 3 * C_);
  transpose_cast_bf16_kernel<<<1024, 256, 0, stream>>>(W_proj, WpT, C_, C_);

  // 2) qkv = x @ W_attn + b_attn   (bf16 out for the attention stage)
  gemm_bf16_nt_kernel<<<dim3(3 * C_ / BN, M_ / BM), 256, 0, stream>>>(
      xb, WaT, b_attn, nullptr, qkvb, M_, 3 * C_, C_);

  // 3) causal flash attention -> y (bf16)
  flash_attn_kernel<<<dim3(T_ / 128, NH_, B_), 256, 0, stream>>>(qkvb, yb);

  // 4) out = y @ W_proj + b_proj   (fp32 final output)
  gemm_bf16_nt_kernel<<<dim3(C_ / BN, M_ / BM), 256, 0, stream>>>(
      yb, WpT, b_proj, out, nullptr, M_, C_, C_);
}
